// EGNNLayer_39298950759104
// MI455X (gfx1250) — matrix operand order repaired and checked by reading, not compiled, not run on hardware
//
#include <hip/hip_runtime.h>
#include <hip/hip_bf16.h>

typedef __attribute__((ext_vector_type(16))) _Float16 v16h;
typedef __attribute__((ext_vector_type(8)))  float    v8f;

#define DIM 128
#define HID 64

// 16-bit A-matrix 16x32 lane layout (ISA 7.12.2):
// lanes 0-15: i=0..7 -> K=i, i=8..15 -> K=16+(i-8); lanes 16-31: +8
__device__ __forceinline__ int klocalA(int i, int lane) {
  return (i < 8 ? i : i + 8) + ((lane & 16) ? 8 : 0);
}
// 16-bit B-matrix 32x16 lane layout (same K order as A):
// lanes 0-15: i=0..7 -> K=i, i=8..15 -> K=16+(i-8); lanes 16-31: +8; col n = lane%16
__device__ __forceinline__ int klocalB(int i, int lane) {
  return (i < 8 ? i : i + 8) + ((lane & 16) ? 8 : 0);
}
__device__ __forceinline__ float silu_(float v)    { return v / (1.f + __expf(-v)); }
__device__ __forceinline__ float sigmoid_(float v) { return 1.f / (1.f + __expf(-v)); }
__device__ __forceinline__ float red16(float v) {   // sum across 16-lane group (wave32)
  v += __shfl_xor(v, 1, 16);
  v += __shfl_xor(v, 2, 16);
  v += __shfl_xor(v, 4, 16);
  v += __shfl_xor(v, 8, 16);
  return v;
}

// f32 -> f16 copy of a dense array (one-time per launch; keeps hot-loop gathers at 2B/elem)
__global__ void prep_f16_kernel(const float* __restrict__ src, _Float16* __restrict__ dst,
                                int total) {
  int t = blockIdx.x * blockDim.x + threadIdx.x;
  if (t < total) dst[t] = (_Float16)src[t];
}

// Convert a row-major f32 weight (N_out x K_in) into WMMA B-fragment order:
// frag[((kt*NT+nt)*32 + lane)*16 + i] = f16( W[nt*16 + lane%16][kt*32 + klocalB(i,lane)] )
__global__ void prep_weight_kernel(const float* __restrict__ W, _Float16* __restrict__ frag,
                                   int K_in, int KT, int NT) {
  int t = blockIdx.x * blockDim.x + threadIdx.x;
  int total = KT * NT * 512;
  if (t >= total) return;
  int i    = t & 15;
  int lane = (t >> 4) & 31;
  int rest = t >> 9;
  int nt   = rest % NT;
  int kt   = rest / NT;
  int n = nt * 16 + (lane & 15);
  int k = kt * 32 + klocalB(i, lane);
  float v = (k < K_in) ? W[(size_t)n * K_in + k] : 0.f;
  frag[t] = (_Float16)v;
}

// ---------------- Edge kernel: one 16-edge tile per wave ----------------
__global__ __launch_bounds__(256) void egnn_edge_kernel(
    const _Float16* __restrict__ hf, const float* __restrict__ x,
    const int* __restrict__ ei, int E_,
    const _Float16* __restrict__ fWe1, const float* __restrict__ be1,
    const _Float16* __restrict__ fWe2, const float* __restrict__ be2,
    const float* __restrict__ ln_g, const float* __restrict__ ln_b,
    const float* __restrict__ Wx, const float* __restrict__ bx,
    const float* __restrict__ Wg, const float* __restrict__ bg,
    float* __restrict__ agg_h, float* __restrict__ agg_x, float* __restrict__ degb)
{
  __shared__ _Float16 lds_z[8][16 * HID];     // per-wave 16x64 f16 staging (16 KB/block)
  const int lane = threadIdx.x & 31;
  const int wv   = threadIdx.x >> 5;
  const int tile = blockIdx.x * 8 + wv;
  const int tilesE = E_ >> 4;
  const bool active = tile < tilesE;
  const int m  = lane & 15;                   // row owned for A / column for B,C
  const int hv = (lane >> 4) & 1;             // which half of the wave

  if (active) {
    const int ebase = tile * 16;
    const int e_m   = ebase + m;
    const int src_m = ei[e_m];
    const int dst_m = ei[E_ + e_m];
    const float rx = x[src_m*3+0] - x[dst_m*3+0];
    const float ry = x[src_m*3+1] - x[dst_m*3+1];
    const float rz = x[src_m*3+2] - x[dst_m*3+2];
    const float dsq = rx*rx + ry*ry + rz*rz;
    const _Float16 dsq_h = (_Float16)dsq;

    // ---- layer 1: (16 x 288pad) x (288 x 64) ----
    v8f acc[4] = {};
    for (int kt = 0; kt < 9; ++kt) {
      v16h a;
      const int k0 = kt * 32;
      #pragma unroll
      for (int i = 0; i < 16; ++i) {
        const int k = k0 + klocalA(i, lane);
        _Float16 v;
        if (k < DIM)          v = hf[(size_t)src_m * DIM + k];
        else if (k < 2*DIM)   v = hf[(size_t)dst_m * DIM + (k - DIM)];
        else if (k == 2*DIM)  v = dsq_h;
        else                  v = (_Float16)0.f;
        a[i] = v;
      }
      #pragma unroll
      for (int nt = 0; nt < 4; ++nt) {
        const v16h b = *(const v16h*)(fWe1 + ((size_t)(kt*4 + nt)*32 + lane)*16);
        acc[nt] = __builtin_amdgcn_wmma_f32_16x16x32_f16(false, a, false, b,
                                                         (short)0, acc[nt], false, false);
      }
    }
    // silu + bias -> LDS (row-major 16x64 f16) for layout change C->A
    #pragma unroll
    for (int nt = 0; nt < 4; ++nt) {
      const int n = nt*16 + m;
      const float bias = be1[n];
      #pragma unroll
      for (int j = 0; j < 8; ++j) {
        const int row = j + hv*8;
        lds_z[wv][row*HID + n] = (_Float16)silu_(acc[nt][j] + bias);
      }
    }
  }
  __syncthreads();
  if (active) {
    const int ebase = tile * 16;
    // ---- layer 2: (16 x 64) x (64 x 64) ----
    v8f acc2[4] = {};
    #pragma unroll
    for (int kt = 0; kt < 2; ++kt) {
      v16h a;
      #pragma unroll
      for (int i = 0; i < 16; ++i) {
        const int k = kt*32 + klocalA(i, lane);
        a[i] = lds_z[wv][m*HID + k];
      }
      #pragma unroll
      for (int nt = 0; nt < 4; ++nt) {
        const v16h b = *(const v16h*)(fWe2 + ((size_t)(kt*4 + nt)*32 + lane)*16);
        acc2[nt] = __builtin_amdgcn_wmma_f32_16x16x32_f16(false, a, false, b,
                                                          (short)0, acc2[nt], false, false);
      }
    }
    // silu + bias, layernorm stats (row m_j = j + 8*hv spans the 16 lanes of this half)
    float val[4][8];
    float s[8], sq[8];
    #pragma unroll
    for (int j = 0; j < 8; ++j) { s[j] = 0.f; sq[j] = 0.f; }
    #pragma unroll
    for (int nt = 0; nt < 4; ++nt) {
      const int n = nt*16 + m;
      const float bias = be2[n];
      #pragma unroll
      for (int j = 0; j < 8; ++j) {
        float v = silu_(acc2[nt][j] + bias);
        val[nt][j] = v; s[j] += v; sq[j] += v*v;
      }
    }
    float mu[8], rstd[8];
    #pragma unroll
    for (int j = 0; j < 8; ++j) {
      float sj = red16(s[j]), qj = red16(sq[j]);
      float mj = sj * (1.f / HID);
      mu[j] = mj;
      rstd[j] = rsqrtf(qj * (1.f / HID) - mj*mj + 1e-5f);
    }
    // affine + gate dot
    float gp[8];
    #pragma unroll
    for (int j = 0; j < 8; ++j) gp[j] = 0.f;
    #pragma unroll
    for (int nt = 0; nt < 4; ++nt) {
      const int n = nt*16 + m;
      const float g = ln_g[n], b = ln_b[n], wg = Wg[n];
      #pragma unroll
      for (int j = 0; j < 8; ++j) {
        float v = (val[nt][j] - mu[j]) * rstd[j] * g + b;
        val[nt][j] = v;
        gp[j] += v * wg;
      }
    }
    float alpha[8], cp[8], cw[8];
    #pragma unroll
    for (int j = 0; j < 8; ++j) { alpha[j] = sigmoid_(red16(gp[j]) + bg[0]); cp[j] = 0.f; }
    #pragma unroll
    for (int nt = 0; nt < 4; ++nt) {
      const int n = nt*16 + m;
      const float wx = Wx[n];
      #pragma unroll
      for (int j = 0; j < 8; ++j) {
        val[nt][j] *= alpha[j];
        cp[j] += val[nt][j] * wx;
      }
    }
    #pragma unroll
    for (int j = 0; j < 8; ++j) cw[j] = tanhf(red16(cp[j]) + bx[0]);

    // scatter gated message: agg_h[dst] += m_gated
    #pragma unroll
    for (int j = 0; j < 8; ++j) {
      const int row = j + hv*8;
      const int dj = ei[E_ + ebase + row];
      #pragma unroll
      for (int nt = 0; nt < 4; ++nt)
        atomicAdd(&agg_h[(size_t)dj*HID + nt*16 + m], val[nt][j]);
    }
    // scatter coordinate update + degree (lanes with m<8 own row m + 8*hv; cw[j=m] valid here)
    if (m < 8) {
      const int row = m + hv*8;
      const int e  = ebase + row;
      const int sr = ei[e], dr = ei[E_ + e];
      float rx = x[sr*3+0] - x[dr*3+0];
      float ry = x[sr*3+1] - x[dr*3+1];
      float rz = x[sr*3+2] - x[dr*3+2];
      float dn = sqrtf(rx*rx + ry*ry + rz*rz) + 1e-8f;
      float c = cw[m] / dn;
      atomicAdd(&agg_x[(size_t)dr*3+0], rx*c);
      atomicAdd(&agg_x[(size_t)dr*3+1], ry*c);
      atomicAdd(&agg_x[(size_t)dr*3+2], rz*c);
      atomicAdd(&degb[dr], 1.f);
    }
  }
}

// ---------------- Node kernel: one 16-node tile per wave ----------------
__global__ __launch_bounds__(256) void egnn_node_kernel(
    const float* __restrict__ h, const _Float16* __restrict__ hf,
    const float* __restrict__ x,
    const float* __restrict__ agg_h, const float* __restrict__ agg_x,
    const float* __restrict__ degb,
    const _Float16* __restrict__ fWh1, const float* __restrict__ bh1,
    const _Float16* __restrict__ fWh2, const float* __restrict__ bh2,
    float* __restrict__ out_h, float* __restrict__ out_x, int N_)
{
  __shared__ _Float16 lds_z[8][16 * HID];
  const int lane = threadIdx.x & 31;
  const int wv   = threadIdx.x >> 5;
  const int tile = blockIdx.x * 8 + wv;
  const int tiles = (N_ + 15) >> 4;
  const bool active = tile < tiles;
  const int m  = lane & 15;
  const int hv = (lane >> 4) & 1;

  if (active) {
    const int nbase = tile * 16;
    const int node_m = nbase + m;
    // ---- layer 1: (16 x 192) x (192 x 64), K = [h(128) | agg_h(64)] ----
    v8f acc[4] = {};
    for (int kt = 0; kt < 6; ++kt) {
      v16h a;
      const int k0 = kt * 32;
      #pragma unroll
      for (int i = 0; i < 16; ++i) {
        const int k = k0 + klocalA(i, lane);
        _Float16 v = (k < DIM) ? hf[(size_t)node_m * DIM + k]
                               : (_Float16)agg_h[(size_t)node_m * HID + (k - DIM)];
        a[i] = v;
      }
      #pragma unroll
      for (int nt = 0; nt < 4; ++nt) {
        const v16h b = *(const v16h*)(fWh1 + ((size_t)(kt*4 + nt)*32 + lane)*16);
        acc[nt] = __builtin_amdgcn_wmma_f32_16x16x32_f16(false, a, false, b,
                                                         (short)0, acc[nt], false, false);
      }
    }
    #pragma unroll
    for (int nt = 0; nt < 4; ++nt) {
      const int n = nt*16 + m;
      const float bias = bh1[n];
      #pragma unroll
      for (int j = 0; j < 8; ++j) {
        const int row = j + hv*8;
        lds_z[wv][row*HID + n] = (_Float16)silu_(acc[nt][j] + bias);
      }
    }
  }
  __syncthreads();
  if (active) {
    const int nbase = tile * 16;
    // ---- layer 2: (16 x 64) x (64 x 128) ----
    v8f acc2[8] = {};
    #pragma unroll
    for (int kt = 0; kt < 2; ++kt) {
      v16h a;
      #pragma unroll
      for (int i = 0; i < 16; ++i) {
        const int k = kt*32 + klocalA(i, lane);
        a[i] = lds_z[wv][m*HID + k];
      }
      #pragma unroll
      for (int nt = 0; nt < 8; ++nt) {
        const v16h b = *(const v16h*)(fWh2 + ((size_t)(kt*8 + nt)*32 + lane)*16);
        acc2[nt] = __builtin_amdgcn_wmma_f32_16x16x32_f16(false, a, false, b,
                                                          (short)0, acc2[nt], false, false);
      }
    }
    // residual h update
    #pragma unroll
    for (int nt = 0; nt < 8; ++nt) {
      const int n = nt*16 + m;
      const float bias = bh2[n];
      #pragma unroll
      for (int j = 0; j < 8; ++j) {
        const int node = nbase + j + hv*8;
        out_h[(size_t)node*DIM + n] = h[(size_t)node*DIM + n] + acc2[nt][j] + bias;
      }
    }
    // coordinate mean-agg + residual (one lane per node)
    if (lane < 16) {
      const int node = nbase + lane;
      const float dg = fmaxf(degb[node], 1.f);
      #pragma unroll
      for (int c = 0; c < 3; ++c)
        out_x[(size_t)node*3 + c] = x[(size_t)node*3 + c] + agg_x[(size_t)node*3 + c] / dg;
    }
  }
}

extern "C" void kernel_launch(void* const* d_in, const int* in_sizes, int n_in,
                              void* d_out, int out_size, void* d_ws, size_t ws_size,
                              hipStream_t stream) {
  const float* h    = (const float*)d_in[0];
  const float* x    = (const float*)d_in[1];
  const int*   ei   = (const int*)  d_in[2];
  const float* We1  = (const float*)d_in[3];
  const float* be1  = (const float*)d_in[4];
  const float* We2  = (const float*)d_in[5];
  const float* be2  = (const float*)d_in[6];
  const float* ln_g = (const float*)d_in[7];
  const float* ln_b = (const float*)d_in[8];
  const float* Wh1  = (const float*)d_in[9];
  const float* bh1  = (const float*)d_in[10];
  const float* Wh2  = (const float*)d_in[11];
  const float* bh2  = (const float*)d_in[12];
  const float* Wx   = (const float*)d_in[13];
  const float* bx   = (const float*)d_in[14];
  const float* Wg   = (const float*)d_in[15];
  const float* bg   = (const float*)d_in[16];
  (void)n_in; (void)out_size; (void)ws_size;

  const int N_ = in_sizes[0] / DIM;   // 50000
  const int E_ = in_sizes[2] / 2;     // 800000

  char* ws = (char*)d_ws;
  size_t off = 0;
  auto carve = [&](size_t bytes) -> void* {
    void* p = ws + off;
    off = (off + bytes + 255) & ~(size_t)255;
    return p;
  };
  float*    agg_h = (float*)   carve((size_t)N_ * HID * 4);
  float*    agg_x = (float*)   carve((size_t)N_ * 3 * 4);
  float*    degb  = (float*)   carve((size_t)N_ * 4);
  _Float16* hf    = (_Float16*)carve((size_t)N_ * DIM * 2);
  _Float16* fWe1  = (_Float16*)carve((size_t)9*4*512*2);   // K=288pad, N=64
  _Float16* fWe2  = (_Float16*)carve((size_t)2*4*512*2);   // K=64,     N=64
  _Float16* fWh1  = (_Float16*)carve((size_t)6*4*512*2);   // K=192,    N=64
  _Float16* fWh2  = (_Float16*)carve((size_t)2*8*512*2);   // K=64,     N=128

  // zero accumulators every launch (d_ws is not re-poisoned between replays)
  hipMemsetAsync(agg_h, 0, (size_t)N_ * HID * 4, stream);
  hipMemsetAsync(agg_x, 0, (size_t)N_ * 3   * 4, stream);
  hipMemsetAsync(degb,  0, (size_t)N_       * 4, stream);

  // h -> f16 copy (feeds all WMMA A-fragment gathers at 2B/element, no cvt in hot loop)
  { int t = N_ * DIM; prep_f16_kernel<<<(t+255)/256, 256, 0, stream>>>(h, hf, t); }

  // weight -> B-fragment swizzle (cheap, once per launch)
  { int t = 9*4*512; prep_weight_kernel<<<(t+255)/256, 256, 0, stream>>>(We1, fWe1, 2*DIM+1, 9, 4); }
  { int t = 2*4*512; prep_weight_kernel<<<(t+255)/256, 256, 0, stream>>>(We2, fWe2, HID,     2, 4); }
  { int t = 6*4*512; prep_weight_kernel<<<(t+255)/256, 256, 0, stream>>>(Wh1, fWh1, DIM+HID, 6, 4); }
  { int t = 2*8*512; prep_weight_kernel<<<(t+255)/256, 256, 0, stream>>>(Wh2, fWh2, HID,     2, 8); }

  const int tilesE = E_ / 16;
  egnn_edge_kernel<<<(tilesE + 7) / 8, 256, 0, stream>>>(
      hf, x, ei, E_, fWe1, be1, fWe2, be2, ln_g, ln_b, Wx, bx, Wg, bg,
      agg_h, agg_x, degb);

  float* out_h = (float*)d_out;
  float* out_x = out_h + (size_t)N_ * DIM;
  const int tilesN = (N_ + 15) / 16;
  egnn_node_kernel<<<(tilesN + 7) / 8, 256, 0, stream>>>(
      h, hf, x, agg_h, agg_x, degb, fWh1, bh1, fWh2, bh2, out_h, out_x, N_);
}
